// WindowAttention_83339545412156
// MI455X (gfx1250) — matrix operand order repaired and checked by reading, not compiled, not run on hardware
//
#include <hip/hip_runtime.h>
#include <hip/hip_fp16.h>
#include <hip/hip_bf16.h>

typedef __attribute__((ext_vector_type(16))) _Float16 v16h;
typedef __attribute__((ext_vector_type(8)))  _Float16 v8h;
typedef __attribute__((ext_vector_type(8)))  float    v8f;

#define WMMA_F16(A, B, C) \
  __builtin_amdgcn_wmma_f32_16x16x32_f16(false, (A), false, (B), (short)0, (C), false, false)

#if __has_builtin(__builtin_amdgcn_global_load_async_to_lds_b128)
#define HAVE_ASYNC_LDS 1
typedef int v4i __attribute__((vector_size(16)));
typedef __attribute__((address_space(1))) v4i g_v4i;   // global (prints as __device__)
typedef __attribute__((address_space(3))) v4i l_v4i;   // LDS
#else
#define HAVE_ASYNC_LDS 0
#endif

__device__ __forceinline__ void wait_async0() {
#if __has_builtin(__builtin_amdgcn_s_wait_asynccnt)
  __builtin_amdgcn_s_wait_asynccnt(0);
#elif defined(__gfx1250__)
  asm volatile("s_wait_asynccnt 0x0" ::: "memory");
#endif
}

// Load a v16h WMMA operand as two 16-byte LDS chunks (lowers to 2x ds_load_b128).
__device__ __forceinline__ v16h ld_pair(const _Float16* lo, const _Float16* hi) {
  v8h l = *(const v8h*)lo;
  v8h h = *(const v8h*)hi;
  v16h r;
#pragma unroll
  for (int i = 0; i < 8; ++i) { r[i] = l[i]; r[i + 8] = h[i]; }
  return r;
}

// ---------------------------------------------------------------------------
// Kernel 1: continuous relative position bias MLP  (N,N,2)->(H,N,N)
// ---------------------------------------------------------------------------
__global__ void bias_kernel(const float* __restrict__ rel,
                            const float* __restrict__ fc1_w, const float* __restrict__ fc1_b,
                            const float* __restrict__ fc2_w, const float* __restrict__ fc2_b,
                            float* __restrict__ bias) {
  int i = blockIdx.x;   // 0..255
  int j = threadIdx.x;  // 0..255
  float r0 = rel[(i * 256 + j) * 2 + 0];
  float r1 = rel[(i * 256 + j) * 2 + 1];
  float acc[4] = {0.f, 0.f, 0.f, 0.f};
  for (int m = 0; m < 128; ++m) {
    float t = r0 * fc1_w[m] + r1 * fc1_w[128 + m] + fc1_b[m];
    // tanh-approximate GELU (approximate=True)
    float g = 0.5f * t * (1.0f + tanhf(0.7978845608f * (t + 0.044715f * t * t * t)));
#pragma unroll
    for (int hh = 0; hh < 4; ++hh) acc[hh] += g * fc2_w[m * 4 + hh];
  }
#pragma unroll
  for (int hh = 0; hh < 4; ++hh)
    bias[hh * 65536 + i * 256 + j] = acc[hh] + fc2_b[hh];
}

// ---------------------------------------------------------------------------
// Kernel 2: qkv = x @ qkv_w + qkv_b   (per window, WMMA), + per-block ssq(q),ssq(k)
// LDS: X[256][128] f16 (64KB) | WT[384][128] f16 (96KB) | red[256] f32 (1KB)
// ---------------------------------------------------------------------------
__global__ void qkv_kernel(const float* __restrict__ x,
                           const float* __restrict__ qkv_w, const float* __restrict__ qkv_b,
                           _Float16* __restrict__ qkv_out,
                           float* __restrict__ partq, float* __restrict__ partk) {
  extern __shared__ char smem[];
  _Float16* X  = (_Float16*)smem;        // [token][c]   stride 128
  _Float16* WT = X + 256 * 128;          // [n][k]       stride 128 (transposed weights)
  float*    red = (float*)(WT + 384 * 128);

  const int tid = threadIdx.x;
  const int b = blockIdx.x;
  const float* xw = x + (size_t)b * 256 * 128;
  for (int idx = tid; idx < 256 * 128; idx += 256) X[idx] = (_Float16)xw[idx];
  for (int idx = tid; idx < 128 * 384; idx += 256) {
    int k = idx / 384, n = idx % 384;
    WT[n * 128 + k] = (_Float16)qkv_w[idx];
  }
  __syncthreads();

  const int lane = tid & 31, wave = tid >> 5;
  const int lg = lane >> 4, ln = lane & 15;
  float ssqq = 0.f, ssqk = 0.f;
  _Float16* outw = qkv_out + (size_t)b * 256 * 384;

  for (int nt = wave * 3; nt < wave * 3 + 3; ++nt) {   // 24 N-tiles / 8 waves
    const int colg = nt * 16 + ln;
    const float cb = qkv_b[colg];
    for (int mt = 0; mt < 16; ++mt) {
      v8f acc = {};
#pragma unroll
      for (int kk = 0; kk < 4; ++kk) {   // K = 128 = 4 x 32
        const _Float16* ap = X + (mt * 16 + ln) * 128 + kk * 32 + lg * 16;
        v16h A = ld_pair(ap, ap + 8);
        const _Float16* bp = WT + colg * 128 + kk * 32 + lg * 16;
        v16h B = ld_pair(bp, bp + 8);
        acc = WMMA_F16(A, B, acc);
      }
#pragma unroll
      for (int r = 0; r < 8; ++r) {
        float v = acc[r] + cb;
        int token = mt * 16 + r + 8 * lg;
        outw[token * 384 + colg] = (_Float16)v;
        if (colg < 128)      ssqq += v * v;   // q columns
        else if (colg < 256) ssqk += v * v;   // k columns
      }
    }
  }
  // deterministic fixed-order block reduction
  red[tid] = ssqq; __syncthreads();
  for (int s = 128; s > 0; s >>= 1) { if (tid < s) red[tid] += red[tid + s]; __syncthreads(); }
  if (tid == 0) partq[b] = red[0];
  __syncthreads();
  red[tid] = ssqk; __syncthreads();
  for (int s = 128; s > 0; s >>= 1) { if (tid < s) red[tid] += red[tid + s]; __syncthreads(); }
  if (tid == 0) partk[b] = red[0];
}

// ---------------------------------------------------------------------------
// Kernel 3: global norm scale = 1 / (||q|| * ||k|| * max(tau, 0.01))
// ---------------------------------------------------------------------------
__global__ void scale_kernel(const float* __restrict__ partq, const float* __restrict__ partk,
                             const float* __restrict__ tau, float* __restrict__ scale_out) {
  __shared__ float rq[256], rk[256];
  int tid = threadIdx.x;
  float sq = 0.f, sk = 0.f;
  for (int i = tid; i < 1024; i += 256) { sq += partq[i]; sk += partk[i]; }
  rq[tid] = sq; rk[tid] = sk; __syncthreads();
  for (int s = 128; s > 0; s >>= 1) {
    if (tid < s) { rq[tid] += rq[tid + s]; rk[tid] += rk[tid + s]; }
    __syncthreads();
  }
  if (tid == 0) {
    float t = fmaxf(tau[0], 0.01f);
    scale_out[0] = 1.0f / (sqrtf(rq[0]) * sqrtf(rk[0]) * t);
  }
}

// ---------------------------------------------------------------------------
// Kernel 4: attention per (window, head). Flash-style online softmax.
// LDS: Q[256][32] | K[256][32] | VT[32][256] (f16, 16KB each) | P: 8 waves x 16x32 f16
// Q/K staged via GLOBAL_LOAD_ASYNC_TO_LDS_B128 (ASYNCcnt) when available.
// ---------------------------------------------------------------------------
__global__ void attn_kernel(const _Float16* __restrict__ qkv,
                            const float* __restrict__ bias, const float* __restrict__ mask,
                            const float* __restrict__ scale_p,
                            _Float16* __restrict__ attnout) {
  extern __shared__ char smem[];
  _Float16* Q  = (_Float16*)smem;          // [token][c], stride 32
  _Float16* Kk = Q + 256 * 32;             // [token][c], stride 32
  _Float16* VT = Kk + 256 * 32;            // [c][token], stride 256 (transposed)
  _Float16* P  = VT + 32 * 256;            // per-wave 16x32 scratch

  const int b = blockIdx.x, h = blockIdx.y;
  const int tid = threadIdx.x, lane = tid & 31, wave = tid >> 5;
  const int lg = lane >> 4, ln = lane & 15;

  const _Float16* src = qkv + (size_t)b * 256 * 384;
#if HAVE_ASYNC_LDS
  {
    const char* s8 = (const char*)src;     // row stride 384*2 = 768 B
    char* q8 = (char*)(void*)Q;
    char* k8 = (char*)(void*)Kk;
    for (int idx = tid; idx < 1024; idx += 256) {   // 256 tokens x 4 chunks of 16B
      int t = idx >> 2, ch = (idx & 3) * 16;
      __builtin_amdgcn_global_load_async_to_lds_b128(
          (g_v4i*)(s8 + t * 768 + h * 64 + ch),
          (l_v4i*)(q8 + t * 64 + ch), 0, 0);
      __builtin_amdgcn_global_load_async_to_lds_b128(
          (g_v4i*)(s8 + t * 768 + 256 + h * 64 + ch),
          (l_v4i*)(k8 + t * 64 + ch), 0, 0);
    }
    for (int idx = tid; idx < 256 * 32; idx += 256) {  // V transposed (manual)
      int t = idx >> 5, c = idx & 31;
      VT[c * 256 + t] = src[t * 384 + 256 + h * 32 + c];
    }
    wait_async0();
  }
#else
  for (int idx = tid; idx < 256 * 32; idx += 256) {
    int t = idx >> 5, c = idx & 31;
    Q[idx]          = src[t * 384 +       h * 32 + c];
    Kk[idx]         = src[t * 384 + 128 + h * 32 + c];
    VT[c * 256 + t] = src[t * 384 + 256 + h * 32 + c];
  }
#endif
  __syncthreads();

  const float scale = scale_p[0];
  const int nw = b & 63;                       // b % NW
  const float* biasH = bias + h * 65536;
  const float* maskW = mask + (size_t)nw * 65536;
  _Float16* Pw = P + wave * 16 * 32;
  _Float16* dst = attnout + (size_t)b * 256 * 128;

  for (int mi = 0; mi < 2; ++mi) {             // 16 M-tiles / 8 waves
    const int mt = wave * 2 + mi;
    const _Float16* ap0 = Q + (mt * 16 + ln) * 32 + lg * 16;
    v16h Aq = ld_pair(ap0, ap0 + 8);

    v8f o0 = {}, o1 = {};
    float rmax[8], rsum[8];
#pragma unroll
    for (int r = 0; r < 8; ++r) { rmax[r] = -1e30f; rsum[r] = 0.f; }

    for (int jp = 0; jp < 8; ++jp) {           // pairs of 16-col S tiles
      v8f s0, s1;
      {
        v8f z = {};
        const _Float16* bp = Kk + ((2 * jp) * 16 + ln) * 32 + lg * 16;
        v16h Bk = ld_pair(bp, bp + 8);
        s0 = WMMA_F16(Aq, Bk, z);              // S tile = q . k^T, K = d = 32
        bp = Kk + ((2 * jp + 1) * 16 + ln) * 32 + lg * 16;
        Bk = ld_pair(bp, bp + 8);
        s1 = WMMA_F16(Aq, Bk, z);
      }
      const int c0 = jp * 32 + ln;
      const int c1 = c0 + 16;
#pragma unroll
      for (int r = 0; r < 8; ++r) {
        const int row = mt * 16 + r + 8 * lg;
        float v0 = s0[r] * scale + biasH[row * 256 + c0] + maskW[row * 256 + c0];
        float v1 = s1[r] * scale + biasH[row * 256 + c1] + maskW[row * 256 + c1];
        float tm = fmaxf(v0, v1);
#pragma unroll
        for (int d_ = 1; d_ < 16; d_ <<= 1) tm = fmaxf(tm, __shfl_xor(tm, d_, 32));
        float nm = fmaxf(rmax[r], tm);
        float alpha = __expf(rmax[r] - nm);
        float p0 = __expf(v0 - nm), p1 = __expf(v1 - nm);
        float ps = p0 + p1;
#pragma unroll
        for (int d_ = 1; d_ < 16; d_ <<= 1) ps += __shfl_xor(ps, d_, 32);
        rsum[r] = rsum[r] * alpha + ps;
        rmax[r] = nm;
        o0[r] *= alpha; o1[r] *= alpha;
        // scatter P into per-wave LDS tile (same-wave LDS ops are in-order)
        Pw[(r + 8 * lg) * 32 + ln]      = (_Float16)p0;
        Pw[(r + 8 * lg) * 32 + 16 + ln] = (_Float16)p1;
      }
      // O += P (16x32) @ V_block (32x32): two WMMAs (d split into 2 N-halves)
      const _Float16* app = Pw + ln * 32 + lg * 16;
      v16h Ap = ld_pair(app, app + 8);
      const _Float16* bv0 = VT + ln * 256        + jp * 32 + lg * 16;
      const _Float16* bv1 = VT + (16 + ln) * 256 + jp * 32 + lg * 16;
      v16h Bv0 = ld_pair(bv0, bv0 + 8);
      v16h Bv1 = ld_pair(bv1, bv1 + 8);
      o0 = WMMA_F16(Ap, Bv0, o0);
      o1 = WMMA_F16(Ap, Bv1, o1);
    }
#pragma unroll
    for (int r = 0; r < 8; ++r) {
      const int token = mt * 16 + r + 8 * lg;
      const float inv = 1.0f / rsum[r];
      dst[token * 128 + h * 32 + ln]      = (_Float16)(o0[r] * inv);
      dst[token * 128 + h * 32 + 16 + ln] = (_Float16)(o1[r] * inv);
    }
  }
}

// ---------------------------------------------------------------------------
// Kernel 5: out = attnout @ proj_w + proj_b  (per window, WMMA), f32 output
// LDS: Y[256][128] f16 (64KB, async-staged) | PT[128][128] f16 (32KB)
// ---------------------------------------------------------------------------
__global__ void proj_kernel(const _Float16* __restrict__ attnout,
                            const float* __restrict__ proj_w, const float* __restrict__ proj_b,
                            float* __restrict__ out) {
  extern __shared__ char smem[];
  _Float16* Y  = (_Float16*)smem;        // [token][c], stride 128
  _Float16* PT = Y + 256 * 128;          // [n][k],     stride 128 (transposed)
  const int b = blockIdx.x, tid = threadIdx.x;

  const _Float16* src = attnout + (size_t)b * 256 * 128;
#if HAVE_ASYNC_LDS
  {
    const char* s8 = (const char*)src;
    char* y8 = (char*)(void*)Y;
    for (int idx = tid; idx < 4096; idx += 256)    // 64KB as 16B chunks
      __builtin_amdgcn_global_load_async_to_lds_b128(
          (g_v4i*)(s8 + idx * 16), (l_v4i*)(y8 + idx * 16), 0, 0);
  }
#else
  for (int idx = tid; idx < 256 * 128; idx += 256) Y[idx] = src[idx];
#endif
  for (int idx = tid; idx < 128 * 128; idx += 256) {
    int k = idx >> 7, n = idx & 127;
    PT[n * 128 + k] = (_Float16)proj_w[idx];
  }
#if HAVE_ASYNC_LDS
  wait_async0();
#endif
  __syncthreads();

  const int lane = tid & 31, wave = tid >> 5;
  const int lg = lane >> 4, ln = lane & 15;
  const int colg = wave * 16 + ln;       // 8 N-tiles / 8 waves
  const float cb = proj_b[colg];
  float* dst = out + (size_t)b * 256 * 128;

  for (int mt = 0; mt < 16; ++mt) {
    v8f acc = {};
#pragma unroll
    for (int kk = 0; kk < 4; ++kk) {
      const _Float16* ap = Y + (mt * 16 + ln) * 128 + kk * 32 + lg * 16;
      v16h A = ld_pair(ap, ap + 8);
      const _Float16* bp = PT + colg * 128 + kk * 32 + lg * 16;
      v16h B = ld_pair(bp, bp + 8);
      acc = WMMA_F16(A, B, acc);
    }
#pragma unroll
    for (int r = 0; r < 8; ++r) {
      const int token = mt * 16 + r + 8 * lg;
      dst[token * 128 + colg] = acc[r] + cb;
    }
  }
}

// ---------------------------------------------------------------------------
// Launch
// ---------------------------------------------------------------------------
extern "C" void kernel_launch(void* const* d_in, const int* in_sizes, int n_in,
                              void* d_out, int out_size, void* d_ws, size_t ws_size,
                              hipStream_t stream) {
  const float* x      = (const float*)d_in[0];
  const float* mask   = (const float*)d_in[1];
  const float* rel    = (const float*)d_in[2];
  const float* qkv_w  = (const float*)d_in[3];
  const float* qkv_b  = (const float*)d_in[4];
  const float* proj_w = (const float*)d_in[5];
  const float* proj_b = (const float*)d_in[6];
  const float* fc1_w  = (const float*)d_in[7];
  const float* fc1_b  = (const float*)d_in[8];
  const float* fc2_w  = (const float*)d_in[9];
  const float* fc2_b  = (const float*)d_in[10];
  const float* tau    = (const float*)d_in[11];

  char* ws = (char*)d_ws;
  float*    scale = (float*)ws;                         // 4 B
  float*    partq = (float*)(ws + 1024);                // 4 KB
  float*    partk = (float*)(ws + 5120);                // 4 KB
  float*    biasT = (float*)(ws + 16384);               // 4*256*256*4 = 1 MB
  _Float16* qkv   = (_Float16*)(ws + 2097152ull);       // 1024*256*384*2 = 192 MB
  _Float16* att   = (_Float16*)(ws + 203423744ull);     // 1024*256*128*2 = 64 MB
  float*    out   = (float*)d_out;

  const int QKV_LDS  = 256 * 128 * 2 + 384 * 128 * 2 + 256 * 4;  // 164864
  const int ATTN_LDS = 3 * 256 * 32 * 2 + 8 * 16 * 32 * 2;       // 57344
  const int PROJ_LDS = 256 * 128 * 2 + 128 * 128 * 2;            // 98304

  (void)hipFuncSetAttribute((const void*)qkv_kernel,
                            hipFuncAttributeMaxDynamicSharedMemorySize, QKV_LDS);
  (void)hipFuncSetAttribute((const void*)proj_kernel,
                            hipFuncAttributeMaxDynamicSharedMemorySize, PROJ_LDS);

  bias_kernel<<<256, 256, 0, stream>>>(rel, fc1_w, fc1_b, fc2_w, fc2_b, biasT);
  qkv_kernel<<<1024, 256, QKV_LDS, stream>>>(x, qkv_w, qkv_b, qkv, partq, partk);
  scale_kernel<<<1, 256, 0, stream>>>(partq, partk, tau, scale);
  attn_kernel<<<dim3(1024, 4), 256, ATTN_LDS, stream>>>(qkv, biasT, mask, scale, att);
  proj_kernel<<<1024, 256, PROJ_LDS, stream>>>(att, proj_w, proj_b, out);
}
